// QLSTM_65481071398101
// MI455X (gfx1250) — compile-verified, matrix operand order checked by reading, not run on hardware
//
#include <hip/hip_runtime.h>
#include <math.h>

// ---------------- types ----------------
typedef __attribute__((ext_vector_type(16))) __bf16 v16bf;
typedef __attribute__((ext_vector_type(8)))  __bf16 v8bf;
typedef __attribute__((ext_vector_type(8)))  float  v8f;

typedef __attribute__((ext_vector_type(4))) unsigned int tdm_v4u;
typedef __attribute__((ext_vector_type(8))) int          tdm_v8i;
typedef __attribute__((ext_vector_type(4))) int          tdm_v4i;

union frag_u { v16bf v; v8bf h[2]; };

#define T_STEPS 1024
#define B_DIM   64
#define D_DIM   256
#define H_DIM   256
#define N_COLS  1024   // 4*H

__device__ __forceinline__ unsigned short f2bf(float x) {
    unsigned int u = __float_as_uint(x);
    unsigned int r = (u + 0x7FFFu + ((u >> 16) & 1u)) >> 16;   // RNE
    return (unsigned short)r;
}
__device__ __forceinline__ float bf2f(unsigned short b) {
    return __uint_as_float(((unsigned int)b) << 16);
}
__device__ __forceinline__ float sigmoidf_(float x) {
    return 1.0f / (1.0f + __expf(-x));
}
__device__ __forceinline__ float tanhf_(float x) {
    return 1.0f - 2.0f / (__expf(2.0f * x) + 1.0f);
}

// ---- TDM: DMA one [64 x 1024] bf16 tile (128 KB) from global into LDS ----
// D# group0: count=1 | lds_addr | global_addr | type=2
// D# group1: data_size=2B, tensor 1024x64, tile 1024x64, row stride 1024
__device__ __forceinline__ void tdm_load_z_tile(const unsigned short* zsrc,
                                                unsigned int lds_byte_off) {
    unsigned long long ga = (unsigned long long)zsrc;
    tdm_v4u g0;
    g0[0] = 1u;                                            // count=1, user descriptor
    g0[1] = lds_byte_off;                                  // lds_addr [63:32]
    g0[2] = (unsigned int)(ga & 0xFFFFFFFFu);              // global_addr lo
    g0[3] = (unsigned int)((ga >> 32) & 0x1FFFFFFu)        // global_addr hi [120:96]
            | (2u << 30);                                  // type=2 (image) [127:126]
    tdm_v8i g1;
    g1[0] = (int)(1u << 16);         // workgroup_mask=0, data_size=1 -> 2 bytes
    g1[1] = (int)(1024u << 16);      // tensor_dim0[15:0] at bits[63:48] = 1024
    g1[2] = (int)(64u << 16);        // tensor_dim0 hi=0; tensor_dim1 lo = 64
    g1[3] = (int)(1024u << 16);      // tensor_dim1 hi=0; tile_dim0 = 1024
    g1[4] = 64;                      // tile_dim1 = 64, tile_dim2 = 0
    g1[5] = 1024;                    // tensor_dim0_stride lo = 1024
    g1[6] = 0;                       // stride hi / tensor_dim1_stride lo
    g1[7] = 0;
    tdm_v4i gz4  = {0, 0, 0, 0};
    tdm_v8i gz8  = {0, 0, 0, 0, 0, 0, 0, 0};
    __builtin_amdgcn_tensor_load_to_lds(g0, g1, gz4, gz4, gz8, 0);
}

// -------- prep: f32 -> bf16 copy of inputs (row-major [T*B, 256]) --------
__global__ void cvt_x_kernel(const float* __restrict__ X,
                             unsigned short* __restrict__ Xb, int n) {
    int i = blockIdx.x * blockDim.x + threadIdx.x;
    if (i < n) Xb[i] = f2bf(X[i]);
}

// -------- prep: pack W slice (k_base..k_base+255) into WMMA B-fragment layout --------
// Fragment F = ((g*16 + nt)*8 + ks); flat = F*512 + lane*16 + j  (bf16 elems)
// ISA 16-bit B layout: lane%16 -> column n; K = ks*32 + j + (j>=8 ? 8 : 0) + (lane>=16 ? 8 : 0)
__global__ void pack_w_kernel(const float* __restrict__ Wf, const float* __restrict__ Wi,
                              const float* __restrict__ Wg, const float* __restrict__ Wo,
                              unsigned short* __restrict__ packed, int k_base) {
    int idx  = blockIdx.x * blockDim.x + threadIdx.x;  // 0 .. 262143
    int j    =  idx        & 15;
    int lane = (idx >> 4)  & 31;
    int ks   = (idx >> 9)  & 7;
    int nt   = (idx >> 12) & 15;
    int g    = (idx >> 16) & 3;
    int n = nt * 16 + (lane & 15);
    int K = ks * 32 + j + ((j >= 8) ? 8 : 0) + ((lane >= 16) ? 8 : 0);
    const float* W = (g == 0) ? Wf : (g == 1) ? Wi : (g == 2) ? Wg : Wo;
    packed[idx] = f2bf(W[(size_t)(k_base + K) * H_DIM + n]);
}

// -------- big parallel GEMM: Z[T*B, 1024](bf16) = Xbf16 @ Wx + bias --------
// one wave per 64x16 tile (4 m-subtiles share each B fragment); K=256 -> 8 wmma k-steps
__global__ void gemm_x_kernel(const unsigned short* __restrict__ Xb,
                              const unsigned short* __restrict__ Wxp,
                              const float* __restrict__ bf_, const float* __restrict__ bi_,
                              const float* __restrict__ bg_, const float* __restrict__ bo_,
                              unsigned short* __restrict__ Z) {
    int wave = (blockIdx.x * blockDim.x + threadIdx.x) >> 5;   // 0..65535
    int lane = threadIdx.x & 31;
    int ln   = lane & 15;
    bool hi  = lane >= 16;

    int ms  = wave >> 6;       // 0..1023  (64-row super tile of T*B)
    int ntg = wave & 63;       // 0..63    (16-col tile of 4H)
    int g   = ntg >> 4;

    const unsigned short* a0 = Xb + (size_t)(ms * 64 + ln) * D_DIM + (hi ? 8 : 0);
    const unsigned short* bp = Wxp + (size_t)ntg * 8 * 512 + lane * 16;

    v8f acc[4] = { {}, {}, {}, {} };
#pragma unroll
    for (int ks = 0; ks < 8; ++ks) {
        frag_u b;
        b.h[0] = *(const v8bf*)(const void*)(bp + (size_t)ks * 512);
        b.h[1] = *(const v8bf*)(const void*)(bp + (size_t)ks * 512 + 8);
#pragma unroll
        for (int m = 0; m < 4; ++m) {
            frag_u a;
            const unsigned short* ar = a0 + (size_t)m * 16 * D_DIM + ks * 32;
            a.h[0] = *(const v8bf*)(const void*)ar;
            a.h[1] = *(const v8bf*)(const void*)(ar + 16);
            acc[m] = __builtin_amdgcn_wmma_f32_16x16x32_bf16(false, a.v, false, b.v,
                                                             (short)0, acc[m], false, false);
        }
    }
    const float* bptr = (g == 0) ? bf_ : (g == 1) ? bi_ : (g == 2) ? bg_ : bo_;
    float bias = bptr[(ntg & 15) * 16 + ln];

#pragma unroll
    for (int m = 0; m < 4; ++m) {
        unsigned short* zb = Z + (size_t)(ms * 64 + m * 16 + (hi ? 8 : 0)) * N_COLS
                             + ntg * 16 + ln;
#pragma unroll
        for (int r = 0; r < 8; ++r) zb[(size_t)r * N_COLS] = f2bf(acc[m][r] + bias);
    }
}

// -------- persistent recurrent scan: one workgroup, 32 waves on one WGP --------
// wave w: nt = w%16 (h column tile), mp = w/16 -> row tiles {2mp, 2mp+1}; all 4 gates.
// Z[t] staged into double-buffered LDS by the Tensor Data Mover, one step ahead.
__global__ void __launch_bounds__(1024, 1)
lstm_scan_kernel(const unsigned short* __restrict__ Whp,
                 const unsigned short* __restrict__ Z,
                 float* __restrict__ out) {
    __shared__ unsigned short zbuf[2][B_DIM * N_COLS];   // 2 x 128 KB bf16 gate pre-acts
    __shared__ unsigned short h_lds[B_DIM * H_DIM];      // 32 KB bf16 hidden state

    int tid  = threadIdx.x;
    int wave = tid >> 5, lane = tid & 31;
    int nt   = wave & 15;
    int mp   = wave >> 4;
    int ln   = lane & 15;
    bool hi  = lane >= 16;

    for (int i = tid; i < B_DIM * H_DIM; i += 1024) h_lds[i] = 0;

    // stage Z[0] via TDM, wait, then barrier makes it visible to all waves
    if (wave == 0) {
        tdm_load_z_tile(Z, (unsigned int)(size_t)&zbuf[0][0]);
        __builtin_amdgcn_s_wait_tensorcnt(0);
    }
    __syncthreads();

    v8f c[2] = { {}, {} };

    float* stacked = out;
    float* hx_out  = out + (size_t)T_STEPS * B_DIM * H_DIM;
    float* cx_out  = hx_out + (size_t)B_DIM * H_DIM;

    for (int t = 0; t < T_STEPS; ++t) {
        // kick off async DMA of next step's Z tile into the other parity buffer;
        // that buffer was last read before the previous step's closing barrier.
        if (wave == 0 && t + 1 < T_STEPS) {
            tdm_load_z_tile(Z + (size_t)(t + 1) * B_DIM * N_COLS,
                            (unsigned int)(size_t)&zbuf[(t + 1) & 1][0]);
        }

        // init accumulators from LDS-staged x-part (bias already folded)
        const unsigned short* zcur = &zbuf[t & 1][0];
        v8f acc[4][2];
#pragma unroll
        for (int g = 0; g < 4; ++g)
#pragma unroll
            for (int m = 0; m < 2; ++m) {
                int m_t = mp * 2 + m;
                const unsigned short* zb = zcur + (m_t * 16 + (hi ? 8 : 0)) * N_COLS
                                           + g * H_DIM + nt * 16 + ln;
                v8f v;
#pragma unroll
                for (int r = 0; r < 8; ++r) v[r] = bf2f(zb[r * N_COLS]);
                acc[g][m] = v;
            }

        // recurrent GEMM: K = 256 -> 8 wmma k-steps, 8 wmma per k-step per wave
#pragma unroll
        for (int ks = 0; ks < 8; ++ks) {
            frag_u a[2];
#pragma unroll
            for (int m = 0; m < 2; ++m) {
                int m_t = mp * 2 + m;
                const unsigned short* ar =
                    h_lds + (size_t)(m_t * 16 + ln) * H_DIM + ks * 32 + (hi ? 8 : 0);
                a[m].h[0] = *(const v8bf*)(const void*)ar;         // ds_load_b128
                a[m].h[1] = *(const v8bf*)(const void*)(ar + 16);  // ds_load_b128
            }
#pragma unroll
            for (int g = 0; g < 4; ++g) {
                frag_u b;
                const unsigned short* bb =
                    Whp + (size_t)((g * 16 + nt) * 8 + ks) * 512 + lane * 16;
                b.h[0] = *(const v8bf*)(const void*)bb;
                b.h[1] = *(const v8bf*)(const void*)(bb + 8);
                acc[g][0] = __builtin_amdgcn_wmma_f32_16x16x32_bf16(
                    false, a[0].v, false, b.v, (short)0, acc[g][0], false, false);
                acc[g][1] = __builtin_amdgcn_wmma_f32_16x16x32_bf16(
                    false, a[1].v, false, b.v, (short)0, acc[g][1], false, false);
            }
        }

        __syncthreads();   // everyone done reading h_lds (and this step's zbuf)

        // gates -> state update -> write h (LDS bf16 + global f32)
#pragma unroll
        for (int m = 0; m < 2; ++m) {
            int m_t = mp * 2 + m;
#pragma unroll
            for (int r = 0; r < 8; ++r) {
                float fv = sigmoidf_(acc[0][m][r]);
                float iv = sigmoidf_(acc[1][m][r]);
                float gv = tanhf_(acc[2][m][r]);
                float ov = sigmoidf_(acc[3][m][r]);
                float cv = fv * c[m][r] + iv * gv;
                c[m][r] = cv;
                float hv = ov * tanhf_(cv);
                int row = m_t * 16 + r + (hi ? 8 : 0);
                int col = nt * 16 + ln;
                h_lds[row * H_DIM + col] = f2bf(hv);
                stacked[(size_t)t * B_DIM * H_DIM + (size_t)row * H_DIM + col] = hv;
                if (t == T_STEPS - 1) {
                    hx_out[row * H_DIM + col] = hv;
                    cx_out[row * H_DIM + col] = cv;
                }
            }
        }

        // next step's Z must have landed before anyone reads it after this barrier
        if (wave == 0) __builtin_amdgcn_s_wait_tensorcnt(0);
        __syncthreads();   // h + staged Z ready for next step
    }
}

// ---------------- launcher ----------------
extern "C" void kernel_launch(void* const* d_in, const int* in_sizes, int n_in,
                              void* d_out, int out_size, void* d_ws, size_t ws_size,
                              hipStream_t stream) {
    const float* X   = (const float*)d_in[0];
    const float* Wf  = (const float*)d_in[1];
    const float* bf_ = (const float*)d_in[2];
    const float* Wi  = (const float*)d_in[3];
    const float* bi_ = (const float*)d_in[4];
    const float* Wg  = (const float*)d_in[5];
    const float* bg_ = (const float*)d_in[6];
    const float* Wo  = (const float*)d_in[7];
    const float* bo_ = (const float*)d_in[8];

    char* ws = (char*)d_ws;
    const size_t z_bytes  = (size_t)T_STEPS * B_DIM * N_COLS * sizeof(unsigned short); // 128 MB
    const size_t xb_bytes = (size_t)T_STEPS * B_DIM * D_DIM * sizeof(unsigned short);  // 32 MB
    const size_t wp_bytes = (size_t)4 * 16 * 8 * 512 * sizeof(unsigned short);         // 512 KB

    unsigned short* Z   = (unsigned short*)ws;
    unsigned short* Xb  = (unsigned short*)(ws + z_bytes);
    unsigned short* Wxp = (unsigned short*)(ws + z_bytes + xb_bytes);
    unsigned short* Whp = (unsigned short*)(ws + z_bytes + xb_bytes + wp_bytes);

    const int n_x = T_STEPS * B_DIM * D_DIM;                 // 16,777,216
    cvt_x_kernel<<<(n_x + 255) / 256, 256, 0, stream>>>(X, Xb, n_x);

    pack_w_kernel<<<262144 / 256, 256, 0, stream>>>(Wf, Wi, Wg, Wo, Wxp, 0);     // x-half of W
    pack_w_kernel<<<262144 / 256, 256, 0, stream>>>(Wf, Wi, Wg, Wo, Whp, D_DIM); // h-half of W

    // 1024 m-supertiles * 64 n-tiles = 65536 waves, 8 waves per 256-thread block
    gemm_x_kernel<<<(1024 * 64) / 8, 256, 0, stream>>>(Xb, Wxp, bf_, bi_, bg_, bo_, Z);

    lstm_scan_kernel<<<1, 1024, 0, stream>>>(Whp, Z, (float*)d_out);
}
